// PoseRefineNet_26096221291163
// MI455X (gfx1250) — matrix operand order repaired and checked by reading, not compiled, not run on hardware
//
#include <hip/hip_runtime.h>
#include <hip/hip_bf16.h>

// ---------------------------------------------------------------------------
// PoseRefineNet forward for MI455X (gfx1250, wave32, WMMA).
//
// All dense contractions (ec fc2, kNN gram, sample-group FCs, attention
// Q/K/V/QK^T/att@V/fc, pointwise convs c5/c6) run through one batched WMMA
// GEMM engine: 128x64x64 tiles, 8 waves/block, each wave 32x32 of output ->
// 8 v_wmma_f32_16x16x32_f16 per LDS stage, f32 accumulation.  Tiles are
// staged to LDS as packed f16 pairs via v_cvt_pk_rtz_f16_f32 with float4
// global loads on the (uniform) interior fast path.  Irregular ops
// (top-k=20, batch-axis batchnorm, column-softmax + row renorm, max-over-k,
// pooling, tiny heads) are VALU kernels.  ~80 GFLOP / ~200MB activations:
// fits in L2 (192MB), latency-bound, so fusion + WMMA density is what counts.
// ---------------------------------------------------------------------------

typedef __attribute__((ext_vector_type(16))) _Float16     v16h;
typedef __attribute__((ext_vector_type(8)))  float        v8f;
typedef __attribute__((ext_vector_type(8)))  unsigned int v8u;

static constexpr int NB   = 16;    // batch
static constexpr int NS   = 500;   // points
static constexpr int NK   = 20;    // kNN
static constexpr int NOBJ = 21;

static constexpr long long TOK = (long long)NB * NS;   // 8000 tokens

constexpr size_t al(size_t x) { return (x + 255) & ~(size_t)255; }

// ---- workspace layout (time-disjoint pools) -------------------------------
static constexpr size_t F4        = 4;
static constexpr size_t SZ_TOK512 = (size_t)TOK * 512 * F4;        // 16.384 MB
static constexpr size_t SZ_SGFC   = (size_t)TOK * NK * 128 * F4;   // 81.92 MB

static constexpr size_t OFF_A   = 0;                       // pool A: sg feats / Q,K,V,Z,U
static constexpr size_t SZ_A    = SZ_SGFC;                 // == 5*SZ_TOK512
static constexpr size_t OFF_B   = OFF_A + al(SZ_A);        // pool B: sg fc1 out / token bufs
static constexpr size_t OFF_XCA = OFF_B;
static constexpr size_t OFF_XCB = OFF_XCA + SZ_TOK512;
static constexpr size_t OFF_PF1 = OFF_XCB + SZ_TOK512;
static constexpr size_t OFF_PF2 = OFF_PF1 + al((size_t)TOK * 192 * F4);
static constexpr size_t OFF_PF3 = OFF_PF2 + al((size_t)TOK * 256 * F4);
static constexpr size_t OFF_C6  = OFF_PF3 + al((size_t)TOK * 448 * F4);
static constexpr size_t OFF_BEND= OFF_C6  + al((size_t)TOK * 1024 * F4);
static constexpr size_t SZ_B    = (OFF_BEND - OFF_B) > SZ_SGFC ? (OFF_BEND - OFF_B) : SZ_SGFC;
static constexpr size_t OFF_G   = OFF_B + al(SZ_B);        // gram / attention (B,S,S)
static constexpr size_t OFF_M   = OFF_G + al((size_t)NB * NS * NS * F4);
// misc
static constexpr size_t OFF_SQ   = OFF_M;
static constexpr size_t OFF_IDX  = OFF_SQ   + al((size_t)TOK * F4);
static constexpr size_t OFF_CMAX = OFF_IDX  + al((size_t)TOK * NK * 4);
static constexpr size_t OFF_CSUM = OFF_CMAX + al((size_t)TOK * F4);
static constexpr size_t OFF_HEC  = OFF_CSUM + al((size_t)TOK * F4);
static constexpr size_t OFF_SG1  = OFF_HEC  + al((size_t)TOK * 64 * F4);
static constexpr size_t OFF_SG2  = OFF_SG1  + al((size_t)TOK * 64 * F4);
static constexpr size_t OFF_EMBT = OFF_SG2  + al((size_t)TOK * 128 * F4);
static constexpr size_t OFF_E1   = OFF_EMBT + al((size_t)TOK * 32 * F4);
static constexpr size_t OFF_E2   = OFF_E1   + al((size_t)TOK * 64 * F4);
static constexpr size_t OFF_AP   = OFF_E2   + al((size_t)TOK * 128 * F4);
static constexpr size_t OFF_HR1  = OFF_AP   + al((size_t)NB * 1024 * F4);
static constexpr size_t OFF_HT1  = OFF_HR1  + al((size_t)NB * 512 * F4);
static constexpr size_t OFF_HR2  = OFF_HT1  + al((size_t)NB * 512 * F4);
static constexpr size_t OFF_HT2  = OFF_HR2  + al((size_t)NB * 128 * F4);
static constexpr size_t OFF_HR3  = OFF_HT2  + al((size_t)NB * 128 * F4);
static constexpr size_t OFF_HT3  = OFF_HR3  + al((size_t)NB * NOBJ * 4 * F4);
static constexpr size_t WS_NEED  = OFF_HT3  + al((size_t)NB * NOBJ * 3 * F4);

static __device__ __forceinline__ unsigned int pk16(float x, float y)
{
    // v_cvt_pk_rtz_f16_f32: builtin returns __fp16x2; we only need the bits.
    return __builtin_bit_cast(unsigned int, __builtin_amdgcn_cvt_pkrtz(x, y));
}

static __device__ __forceinline__ v8f wmma_step(v8u a, v8u b, v8f c)
{
    return __builtin_amdgcn_wmma_f32_16x16x32_f16(
        false, __builtin_bit_cast(v16h, a), false, __builtin_bit_cast(v16h, b),
        (short)0, c, false, false);
}

// ---------------------------------------------------------------------------
// Generic batched GEMM: C = A(MxK) * B(KxN) [+bias] [relu], f16 WMMA, f32 acc.
// transB=1 means B is stored (N,K) row-major (weights (Cout,Cin), or K^T).
// Tile: BM=128, BN=64, BK=64.  8 waves: 4x2 grid, each wave 32x32 output.
// LDS stores f16 pairs packed into uints, K-major, row stride 36 uints
// (16B-aligned, conflict-free under wave32 column access).
// ---------------------------------------------------------------------------
__global__ __launch_bounds__(256)
void k_gemm(const float* __restrict__ A, const float* __restrict__ Bm,
            const float* __restrict__ bias, float* __restrict__ C,
            int M, int N, int K, int lda, int ldb, int ldc,
            long long sA, long long sB, long long sC,
            int transB, int relu)
{
    A  += (long long)blockIdx.z * sA;
    Bm += (long long)blockIdx.z * sB;
    C  += (long long)blockIdx.z * sC;

    __shared__ unsigned int As[128][36];   // [m][k/2] : 32 uints + 4 pad
    __shared__ unsigned int Bs[64][36];    // [n][k/2]

    const int tid  = threadIdx.x;
    const int lane = tid & 31;
    const int wave = tid >> 5;
    const int wm   = wave >> 1;            // 0..3 : rows wm*32
    const int wn   = wave & 1;             // 0..1 : cols wn*32
    const int half = lane >> 4;            // K-half select per ISA layout
    const int l16  = lane & 15;

    const int m0 = blockIdx.y * 128;
    const int n0 = blockIdx.x * 64;

    v8f acc00 = {}, acc01 = {}, acc10 = {}, acc11 = {};

    for (int k0 = 0; k0 < K; k0 += 64) {
        const bool fullK = (k0 + 64 <= K);

        // ---- stage A tile ----
        if (fullK && (m0 + 128 <= M)) {
            if (k0 + 64 < K)
                __builtin_prefetch(A + (long long)(m0 + (tid >> 1)) * lda + k0 + 64, 0, 1);
            for (int t = tid; t < 128 * 16; t += 256) {
                int r = t >> 4, q = t & 15;
                const float4 v = *(const float4*)(A + (long long)(m0 + r) * lda + k0 + q * 4);
                As[r][q * 2 + 0] = pk16(v.x, v.y);
                As[r][q * 2 + 1] = pk16(v.z, v.w);
            }
        } else {
            for (int t = tid; t < 128 * 32; t += 256) {
                int r = t >> 5, q = t & 31;
                int gm = m0 + r, gk = k0 + q * 2;
                float x0 = 0.f, x1 = 0.f;
                if (gm < M) {
                    if (gk + 0 < K) x0 = A[(long long)gm * lda + gk + 0];
                    if (gk + 1 < K) x1 = A[(long long)gm * lda + gk + 1];
                }
                As[r][q] = pk16(x0, x1);
            }
        }

        // ---- stage B tile into [n][k] ----
        if (transB) {
            if (fullK && (n0 + 64 <= N)) {
                for (int t = tid; t < 64 * 16; t += 256) {
                    int r = t >> 4, q = t & 15;
                    const float4 v = *(const float4*)(Bm + (long long)(n0 + r) * ldb + k0 + q * 4);
                    Bs[r][q * 2 + 0] = pk16(v.x, v.y);
                    Bs[r][q * 2 + 1] = pk16(v.z, v.w);
                }
            } else {
                for (int t = tid; t < 64 * 32; t += 256) {
                    int r = t >> 5, q = t & 31;
                    int gn = n0 + r, gk = k0 + q * 2;
                    float x0 = 0.f, x1 = 0.f;
                    if (gn < N) {
                        if (gk + 0 < K) x0 = Bm[(long long)gn * ldb + gk + 0];
                        if (gk + 1 < K) x1 = Bm[(long long)gn * ldb + gk + 1];
                    }
                    Bs[r][q] = pk16(x0, x1);
                }
            }
        } else {
            if (fullK && (n0 + 64 <= N)) {
                for (int t = tid; t < 64 * 16; t += 256) {
                    int k = t >> 4, n4 = (t & 15) * 4;
                    const float4 v = *(const float4*)(Bm + (long long)(k0 + k) * ldb + n0 + n4);
                    ((_Float16*)&Bs[n4 + 0][0])[k] = (_Float16)v.x;
                    ((_Float16*)&Bs[n4 + 1][0])[k] = (_Float16)v.y;
                    ((_Float16*)&Bs[n4 + 2][0])[k] = (_Float16)v.z;
                    ((_Float16*)&Bs[n4 + 3][0])[k] = (_Float16)v.w;
                }
            } else {
                for (int t = tid; t < 64 * 64; t += 256) {
                    int k = t >> 6, n = t & 63;
                    int gn = n0 + n, gk = k0 + k;
                    float x = 0.f;
                    if (gn < N && gk < K) x = Bm[(long long)gk * ldb + gn];
                    ((_Float16*)&Bs[n][0])[k] = (_Float16)x;
                }
            }
        }
        __syncthreads();

        // ---- two 32-K subchunks, 4 WMMA each ----
#pragma unroll
        for (int ks = 0; ks < 2; ++ks) {
            const int kb = ks * 16;   // uint base within row
            // A fragments (ISA 7.12.2, 16-bit A 16x32)
            v8u a0, a1;
            {
                const unsigned int* ar0 = &As[wm * 32 + l16][kb];
                const unsigned int* ar1 = &As[wm * 32 + 16 + l16][kb];
#pragma unroll
                for (int r = 0; r < 8; ++r) {
                    int off = ((r >> 2) * 8) + half * 4 + (r & 3);
                    a0[r] = ar0[off];
                    a1[r] = ar1[off];
                }
            }
            // B fragments: lane -> column, same K order as A fragments
            v8u b0, b1;
            {
                const unsigned int* br0 = &Bs[wn * 32 + l16][kb];
                const unsigned int* br1 = &Bs[wn * 32 + 16 + l16][kb];
#pragma unroll
                for (int i = 0; i < 8; ++i) { int off = ((i >> 2) * 8) + half * 4 + (i & 3); b0[i] = br0[off]; b1[i] = br1[off]; }
            }
            acc00 = wmma_step(a0, b0, acc00);
            acc01 = wmma_step(a0, b1, acc01);
            acc10 = wmma_step(a1, b0, acc10);
            acc11 = wmma_step(a1, b1, acc11);
        }
        __syncthreads();
    }

    // epilogue: C frag VGPR r -> M = 8*half + r, N = l16 (per 16x16 tile)
#pragma unroll
    for (int r = 0; r < 8; ++r) {
        int mA = m0 + wm * 32 + half * 8 + r;        // frag row group 0
        int mB = mA + 16;                            // frag row group 1
        int nA = n0 + wn * 32 + l16;                 // frag col group 0
        int nB = nA + 16;                            // frag col group 1
        float bA = bias ? bias[nA < N ? nA : 0] : 0.f;
        float bB = bias ? bias[nB < N ? nB : 0] : 0.f;
        if (mA < M) {
            if (nA < N) {
                float v = acc00[r] + bA;
                if (relu) v = fmaxf(v, 0.f);
                C[(long long)mA * ldc + nA] = v;
            }
            if (nB < N) {
                float v = acc01[r] + bB;
                if (relu) v = fmaxf(v, 0.f);
                C[(long long)mA * ldc + nB] = v;
            }
        }
        if (mB < M) {
            if (nA < N) {
                float v = acc10[r] + bA;
                if (relu) v = fmaxf(v, 0.f);
                C[(long long)mB * ldc + nA] = v;
            }
            if (nB < N) {
                float v = acc11[r] + bB;
                if (relu) v = fmaxf(v, 0.f);
                C[(long long)mB * ldc + nB] = v;
            }
        }
    }
}

// ---------------------------------------------------------------------------
// BatchNorm over (batch, inner) per s; optional relu and residual.  In place.
// ---------------------------------------------------------------------------
__global__ __launch_bounds__(256)
void k_bn(float* __restrict__ X, const float* __restrict__ resid,
          const float* __restrict__ gamma, const float* __restrict__ beta,
          int L, long long strideB, int relu)
{
    const int s = blockIdx.x;
    const int tid = threadIdx.x;
    __shared__ float r1[256], r2[256];
    const long long base = (long long)s * L;
    float sum = 0.f, sq = 0.f;
    for (int b = 0; b < NB; ++b) {
        const float* p = X + b * strideB + base;
        for (int i = tid; i < L; i += 256) { float v = p[i]; sum += v; sq += v * v; }
    }
    r1[tid] = sum; r2[tid] = sq; __syncthreads();
    for (int off = 128; off > 0; off >>= 1) {
        if (tid < off) { r1[tid] += r1[tid + off]; r2[tid] += r2[tid + off]; }
        __syncthreads();
    }
    const float cnt  = (float)NB * (float)L;
    const float mean = r1[0] / cnt;
    const float var  = r2[0] / cnt - mean * mean;
    const float scl  = rsqrtf(var + 1e-5f) * gamma[s];
    const float sh   = beta[s];
    for (int b = 0; b < NB; ++b) {
        float* p = X + b * strideB + base;
        const float* rp = resid ? resid + b * strideB + base : nullptr;
        for (int i = tid; i < L; i += 256) {
            float v = (p[i] - mean) * scl + sh;
            if (relu) v = fmaxf(v, 0.f);
            if (rp) v += rp[i];
            p[i] = v;
        }
    }
}

// ---- kNN: squared norms, top-k selection (one wave32 per (b,s) row) -------
__global__ void k_sq(const float* __restrict__ h, float* __restrict__ sq, int C)
{
    int i = blockIdx.x * blockDim.x + threadIdx.x;
    if (i >= NB * NS) return;
    const float* p = h + (long long)i * C;
    float s = 0.f;
    for (int c = 0; c < C; ++c) { float v = p[c]; s += v * v; }
    sq[i] = s;
}

__global__ void k_topk(const float* __restrict__ gram, const float* __restrict__ sq,
                       int* __restrict__ idx)
{
    const int bs = blockIdx.x;               // b*NS + s
    const int b = bs / NS, s = bs % NS;
    const int lane = threadIdx.x & 31;
    const float* grow = gram + (long long)b * NS * NS + (long long)s * NS;
    const float* sqb  = sq + (long long)b * NS;
    const float  sqs  = sqb[s];

    float dv[16]; int dt[16];
#pragma unroll
    for (int i = 0; i < 16; ++i) { dv[i] = 3.0e38f; dt[i] = 0x7fffffff; }
    for (int i = 0; i < 16; ++i) {
        int t = lane + i * 32;
        if (t < NS) { dv[i] = sqs + sqb[t] - 2.f * grow[t]; dt[i] = t; }
    }
    int* out = idx + (long long)bs * NK;
    for (int j = 0; j < NK; ++j) {
        float bv = 3.0e38f; int bslot = 0;
#pragma unroll
        for (int i = 0; i < 16; ++i)
            if (dv[i] < bv) { bv = dv[i]; bslot = i; }
        float wv = bv; int wt = dt[bslot]; int wl = lane; int ws = bslot;
        for (int off = 16; off > 0; off >>= 1) {
            float ov = __shfl_xor(wv, off);
            int   ot = __shfl_xor(wt, off);
            int   ol = __shfl_xor(wl, off);
            int   os = __shfl_xor(ws, off);
            if (ov < wv || (ov == wv && ot < wt)) { wv = ov; wt = ot; wl = ol; ws = os; }
        }
        if (lane == 0) out[j] = wt;
        if (lane == wl) dv[ws] = 3.0e38f;
    }
}

// feat[b,s,j,0..2C) = concat(h[b,s], h[b,s]-h[b,idx[b,s,j]])
__global__ void k_feat(const float* __restrict__ h, const int* __restrict__ idx,
                       float* __restrict__ feat, int C)
{
    long long i = (long long)blockIdx.x * blockDim.x + threadIdx.x;
    long long total = TOK * NK * 2 * C;
    if (i >= total) return;
    int c = (int)(i % (2 * C));
    long long r = i / (2 * C);
    int j = (int)(r % NK);
    long long bs = r / NK;
    const float* hc = h + bs * C;
    float v;
    if (c < C) v = hc[c];
    else {
        long long b = bs / NS;
        int t = idx[bs * NK + j];
        v = hc[c - C] - h[(b * NS + t) * C + (c - C)];
    }
    feat[i] = v;
}

__global__ void k_maxk(const float* __restrict__ in, float* __restrict__ out, int C)
{
    long long i = (long long)blockIdx.x * blockDim.x + threadIdx.x;
    if (i >= TOK * C) return;
    int c = (int)(i % C);
    long long bs = i / C;
    const float* p = in + bs * NK * C + c;
    float m = p[0];
    for (int j = 1; j < NK; ++j) m = fmaxf(m, p[(long long)j * C]);
    out[i] = m;
}

// ---- offset-attention softmax: softmax over axis 1 (columns), then rows
// renormalized by their sum -------------------------------------------------
__global__ __launch_bounds__(256)
void k_att_col(const float* __restrict__ att, float* __restrict__ cmax,
               float* __restrict__ csum)
{
    int bt = blockIdx.x;                 // b*NS + t
    int b = bt / NS, t = bt % NS;
    const float* base = att + (long long)b * NS * NS + t;
    __shared__ float red[256];
    int tid = threadIdx.x;
    float m = -3.0e38f;
    for (int s = tid; s < NS; s += 256) m = fmaxf(m, base[(long long)s * NS]);
    red[tid] = m; __syncthreads();
    for (int off = 128; off > 0; off >>= 1) {
        if (tid < off) red[tid] = fmaxf(red[tid], red[tid + off]);
        __syncthreads();
    }
    m = red[0];
    __syncthreads();
    float sum = 0.f;
    for (int s = tid; s < NS; s += 256) sum += __expf(base[(long long)s * NS] - m);
    red[tid] = sum; __syncthreads();
    for (int off = 128; off > 0; off >>= 1) {
        if (tid < off) red[tid] += red[tid + off];
        __syncthreads();
    }
    if (tid == 0) { cmax[bt] = m; csum[bt] = red[0]; }
}

__global__ __launch_bounds__(256)
void k_att_row(float* __restrict__ att, const float* __restrict__ cmax,
               const float* __restrict__ csum)
{
    int bs = blockIdx.x;                 // b*NS + s
    int b = bs / NS, s = bs % NS;
    float* row = att + (long long)b * NS * NS + (long long)s * NS;
    const float* cm = cmax + (long long)b * NS;
    const float* cs = csum + (long long)b * NS;
    __shared__ float red[256];
    int tid = threadIdx.x;
    float sum = 0.f;
    for (int t = tid; t < NS; t += 256) {
        float e = __expf(row[t] - cm[t]) / cs[t];
        row[t] = e; sum += e;
    }
    red[tid] = sum; __syncthreads();
    for (int off = 128; off > 0; off >>= 1) {
        if (tid < off) red[tid] += red[tid + off];
        __syncthreads();
    }
    float inv = 1.0f / red[0];
    for (int t = tid; t < NS; t += 256) row[t] *= inv;
}

// ---- small elementwise helpers --------------------------------------------
__global__ void k_ec1(const float* __restrict__ x, const float* __restrict__ W,
                      const float* __restrict__ bias, float* __restrict__ out)
{
    long long i = (long long)blockIdx.x * blockDim.x + threadIdx.x;
    if (i >= TOK * 64) return;
    int o = (int)(i & 63);
    long long bs = i >> 6;
    const float* xp = x + bs * 3;
    out[i] = xp[0] * W[o] + xp[1] * W[64 + o] + xp[2] * W[128 + o] + bias[o];
}

__global__ void k_embT(const float* __restrict__ emb, float* __restrict__ out)
{
    long long i = (long long)blockIdx.x * blockDim.x + threadIdx.x;
    if (i >= TOK * 32) return;
    int c = (int)(i & 31);
    long long bs = i >> 5;
    long long b = bs / NS, s = bs % NS;
    out[i] = emb[(b * 32 + c) * NS + s];
}

__global__ void k_copyc(const float* __restrict__ src, float* __restrict__ dst,
                        int Cs, int Cd, int off)
{
    long long i = (long long)blockIdx.x * blockDim.x + threadIdx.x;
    if (i >= TOK * Cs) return;
    int c = (int)(i % Cs);
    long long bs = i / Cs;
    dst[bs * Cd + off + c] = src[i];
}

__global__ void k_sub(const float* __restrict__ a, const float* __restrict__ b,
                      float* __restrict__ c, long long n)
{
    long long i = (long long)blockIdx.x * blockDim.x + threadIdx.x;
    if (i < n) c[i] = a[i] - b[i];
}

__global__ void k_meanpool(const float* __restrict__ x, float* __restrict__ ap)
{
    int i = blockIdx.x * blockDim.x + threadIdx.x;   // b*1024 + c
    if (i >= NB * 1024) return;
    int c = i & 1023, b = i >> 10;
    const float* p = x + (long long)b * NS * 1024 + c;
    float s = 0.f;
    for (int t = 0; t < NS; ++t) s += p[(long long)t * 1024];
    ap[i] = s * (1.0f / NS);
}

__global__ void k_fc(const float* __restrict__ X, const float* __restrict__ W,
                     const float* __restrict__ bias, float* __restrict__ Y,
                     int K, int N, int relu)
{
    int i = blockIdx.x * blockDim.x + threadIdx.x;   // m*N + n
    if (i >= NB * N) return;
    int n = i % N, m = i / N;
    const float* xp = X + (long long)m * K;
    float s = bias ? bias[n] : 0.f;
    for (int k = 0; k < K; ++k) s += xp[k] * W[(long long)k * N + n];
    if (relu) s = fmaxf(s, 0.f);
    Y[i] = s;
}

__global__ void k_out(const float* __restrict__ r3, const float* __restrict__ t3,
                      const int* __restrict__ obj, float* __restrict__ out)
{
    if (threadIdx.x == 0 && blockIdx.x == 0) {
        int o = obj[0];
        for (int i = 0; i < 4; ++i) out[i] = r3[o * 4 + i];
        for (int i = 0; i < 3; ++i) out[4 + i] = t3[o * 3 + i];
    }
}

// ---------------------------------------------------------------------------
// Host-side orchestration
// ---------------------------------------------------------------------------
static inline void launch_gemm(const float* A, const float* Bm, const float* bias,
                               float* C, int M, int N, int K, int lda, int ldb, int ldc,
                               long long sA, long long sB, long long sC, int batch,
                               int transB, int relu, hipStream_t st)
{
    dim3 g((N + 63) / 64, (M + 127) / 128, batch);
    k_gemm<<<g, 256, 0, st>>>(A, Bm, bias, C, M, N, K, lda, ldb, ldc,
                              sA, sB, sC, transB, relu);
}

static inline dim3 ew(long long n) { return dim3((unsigned)((n + 255) / 256)); }

static void run_sg(const float* h, int C, int Cout,
                   const float* w1, const float* b1, const float* w2, const float* b2,
                   const float* bg, const float* bb,
                   float* sqb, int* idxb, float* G,
                   float* featA, float* fc1B, float* outBuf, hipStream_t st)
{
    k_sq<<<ew(TOK), 256, 0, st>>>(h, sqb, C);
    // gram = h @ h^T per batch
    launch_gemm(h, h, nullptr, G, NS, NS, C, C, C, NS,
                (long long)NS * C, (long long)NS * C, (long long)NS * NS, NB, 1, 0, st);
    k_topk<<<(unsigned)TOK, 32, 0, st>>>(G, sqb, idxb);
    long long nf = TOK * NK * 2 * C;
    k_feat<<<ew(nf), 256, 0, st>>>(h, idxb, featA, C);
    const int Mr = (int)(TOK * NK);
    launch_gemm(featA, w1, b1, fc1B, Mr, Cout, 2 * C, 2 * C, Cout, Cout,
                0, 0, 0, 1, 0, 0, st);
    k_bn<<<NS, 256, 0, st>>>(fc1B, nullptr, bg, bb, NK * Cout,
                             (long long)NS * NK * Cout, 1);
    launch_gemm(fc1B, w2, b2, featA, Mr, Cout, Cout, Cout, Cout, Cout,
                0, 0, 0, 1, 0, 0, st);
    k_bn<<<NS, 256, 0, st>>>(featA, nullptr, bg, bb, NK * Cout,
                             (long long)NS * NK * Cout, 1);
    k_maxk<<<ew(TOK * Cout), 256, 0, st>>>(featA, outBuf, Cout);
}

static void run_oa(const float* xc, float* yOut, int d,
                   const float* qw, const float* qb, const float* kw, const float* kb,
                   const float* vw, const float* vb, const float* fw, const float* fb,
                   const float* bg, const float* bb,
                   float* Q, float* Kb_, float* V, float* Z, float* U, float* G,
                   float* cmax, float* csum, hipStream_t st)
{
    const int M = (int)TOK;
    launch_gemm(xc, qw, qb, Q,   M, d, d, d, d, d, 0, 0, 0, 1, 0, 0, st);
    launch_gemm(xc, kw, kb, Kb_, M, d, d, d, d, d, 0, 0, 0, 1, 0, 0, st);
    launch_gemm(xc, vw, vb, V,   M, d, d, d, d, d, 0, 0, 0, 1, 0, 0, st);
    // QK^T per batch
    launch_gemm(Q, Kb_, nullptr, G, NS, NS, d, d, d, NS,
                (long long)NS * d, (long long)NS * d, (long long)NS * NS, NB, 1, 0, st);
    k_att_col<<<(unsigned)TOK, 256, 0, st>>>(G, cmax, csum);
    k_att_row<<<(unsigned)TOK, 256, 0, st>>>(G, cmax, csum);
    // z = att @ V per batch
    launch_gemm(G, V, nullptr, Z, NS, d, NS, NS, d, d,
                (long long)NS * NS, (long long)NS * d, (long long)NS * d, NB, 0, 0, st);
    long long n = (long long)M * d;
    k_sub<<<ew(n), 256, 0, st>>>(Z, xc, U, n);
    launch_gemm(U, fw, fb, yOut, M, d, d, d, d, d, 0, 0, 0, 1, 0, 0, st);
    // y = relu(bn(...)) + xc
    k_bn<<<NS, 256, 0, st>>>(yOut, xc, bg, bb, d, (long long)NS * d, 1);
}

extern "C" void kernel_launch(void* const* d_in, const int* in_sizes, int n_in,
                              void* d_out, int out_size, void* d_ws, size_t ws_size,
                              hipStream_t stream)
{
    (void)in_sizes; (void)n_in; (void)out_size; (void)ws_size;
#define PF(i) ((const float*)d_in[(i)])
    const float* x   = PF(0);
    const float* emb = PF(1);
    const int*   obj = (const int*)d_in[2];

    char* ws = (char*)d_ws;
    auto W = [ws](size_t off) { return (float*)(ws + off); };

    float* Qb   = W(OFF_A);
    float* Kb   = W(OFF_A + 1 * SZ_TOK512);
    float* Vb   = W(OFF_A + 2 * SZ_TOK512);
    float* Zb   = W(OFF_A + 3 * SZ_TOK512);
    float* Ub   = W(OFF_A + 4 * SZ_TOK512);
    float* featA = W(OFF_A);        // sg phase reuse of pool A
    float* sgfc1 = W(OFF_B);        // sg phase reuse of pool B
    float* xcA  = W(OFF_XCA);
    float* xcB  = W(OFF_XCB);
    float* pf1  = W(OFF_PF1);
    float* pf2  = W(OFF_PF2);
    float* pf3  = W(OFF_PF3);
    float* c6o  = W(OFF_C6);
    float* G    = W(OFF_G);
    float* sqb  = W(OFF_SQ);
    int*   idxb = (int*)(ws + OFF_IDX);
    float* cmax = W(OFF_CMAX);
    float* csum = W(OFF_CSUM);
    float* hec  = W(OFF_HEC);
    float* sg1o = W(OFF_SG1);
    float* sg2o = W(OFF_SG2);
    float* embT = W(OFF_EMBT);
    float* e1   = W(OFF_E1);
    float* e2   = W(OFF_E2);
    float* ap   = W(OFF_AP);
    float* hr1  = W(OFF_HR1);
    float* ht1  = W(OFF_HT1);
    float* hr2  = W(OFF_HR2);
    float* ht2  = W(OFF_HT2);
    float* hr3  = W(OFF_HR3);
    float* ht3  = W(OFF_HT3);

    // ---- encoder: 3 -> 64 -> 64 with per-s batchnorm ----
    k_ec1<<<ew(TOK * 64), 256, 0, stream>>>(x, PF(3), PF(4), xcA);
    k_bn<<<NS, 256, 0, stream>>>(xcA, nullptr, PF(7), PF(8), 64, (long long)NS * 64, 1);
    launch_gemm(xcA, PF(5), PF(6), hec, (int)TOK, 64, 64, 64, 64, 64,
                0, 0, 0, 1, 0, 0, stream);
    k_bn<<<NS, 256, 0, stream>>>(hec, nullptr, PF(7), PF(8), 64, (long long)NS * 64, 1);

    // ---- sample-group blocks ----
    run_sg(hec, 64, 64, PF(9), PF(10), PF(11), PF(12), PF(13), PF(14),
           sqb, idxb, G, featA, sgfc1, sg1o, stream);
    run_sg(sg1o, 64, 128, PF(15), PF(16), PF(17), PF(18), PF(19), PF(20),
           sqb, idxb, G, featA, sgfc1, sg2o, stream);

    // ---- offset attention 1 (d=128) ----
    run_oa(sg2o, xcA, 128, PF(41), PF(42), PF(43), PF(44), PF(45), PF(46),
           PF(47), PF(48), PF(49), PF(50), Qb, Kb, Vb, Zb, Ub, G, cmax, csum, stream);

    // e1 = relu(pw(emb, ec1))  (emb (B,32,S) -> token-major then GEMM transB)
    k_embT<<<ew(TOK * 32), 256, 0, stream>>>(emb, embT);
    launch_gemm(embT, PF(21), PF(22), e1, (int)TOK, 64, 32, 32, 32, 64,
                0, 0, 0, 1, 1, 1, stream);

    // pf1 = concat(xcA[128], e1[64])
    k_copyc<<<ew(TOK * 128), 256, 0, stream>>>(xcA, pf1, 128, 192, 0);
    k_copyc<<<ew(TOK * 64),  256, 0, stream>>>(e1,  pf1, 64, 192, 128);

    // ---- offset attention 2 (d=128) ----
    run_oa(xcA, xcB, 128, PF(51), PF(52), PF(53), PF(54), PF(55), PF(56),
           PF(57), PF(58), PF(59), PF(60), Qb, Kb, Vb, Zb, Ub, G, cmax, csum, stream);

    // e2 = relu(pw(e1, ec2))
    launch_gemm(e1, PF(23), PF(24), e2, (int)TOK, 128, 64, 64, 64, 128,
                0, 0, 0, 1, 1, 1, stream);

    // pf2 = concat(xcB[128], e2[128]); pf3 = concat(pf1, pf2)
    k_copyc<<<ew(TOK * 128), 256, 0, stream>>>(xcB, pf2, 128, 256, 0);
    k_copyc<<<ew(TOK * 128), 256, 0, stream>>>(e2,  pf2, 128, 256, 128);
    k_copyc<<<ew(TOK * 192), 256, 0, stream>>>(pf1, pf3, 192, 448, 0);
    k_copyc<<<ew(TOK * 256), 256, 0, stream>>>(pf2, pf3, 256, 448, 192);

    // ---- offset attention 3 (d=448) ----
    run_oa(pf3, xcA, 448, PF(61), PF(62), PF(63), PF(64), PF(65), PF(66),
           PF(67), PF(68), PF(69), PF(70), Qb, Kb, Vb, Zb, Ub, G, cmax, csum, stream);

    // c5: 448 -> 512 (+relu)
    launch_gemm(xcA, PF(25), PF(26), xcB, (int)TOK, 512, 448, 448, 448, 512,
                0, 0, 0, 1, 1, 1, stream);

    // ---- offset attention 4 (d=512) ----
    run_oa(xcB, xcA, 512, PF(71), PF(72), PF(73), PF(74), PF(75), PF(76),
           PF(77), PF(78), PF(79), PF(80), Qb, Kb, Vb, Zb, Ub, G, cmax, csum, stream);

    // c6: 512 -> 1024 (+relu)
    launch_gemm(xcA, PF(27), PF(28), c6o, (int)TOK, 1024, 512, 512, 512, 1024,
                0, 0, 0, 1, 1, 1, stream);

    // mean pool + heads
    k_meanpool<<<ew(NB * 1024), 256, 0, stream>>>(c6o, ap);
    k_fc<<<ew(NB * 512), 256, 0, stream>>>(ap,  PF(29), PF(30), hr1, 1024, 512, 1);
    k_fc<<<ew(NB * 512), 256, 0, stream>>>(ap,  PF(31), PF(32), ht1, 1024, 512, 1);
    k_fc<<<ew(NB * 128), 256, 0, stream>>>(hr1, PF(33), PF(34), hr2, 512, 128, 1);
    k_fc<<<ew(NB * 128), 256, 0, stream>>>(ht1, PF(35), PF(36), ht2, 512, 128, 1);
    k_fc<<<ew(NB * NOBJ * 4), 256, 0, stream>>>(hr2, PF(37), PF(38), hr3, 128, NOBJ * 4, 0);
    k_fc<<<ew(NB * NOBJ * 3), 256, 0, stream>>>(ht2, PF(39), PF(40), ht3, 128, NOBJ * 3, 0);

    k_out<<<1, 1, 0, stream>>>(hr3, ht3, obj, (float*)d_out);
#undef PF
}
